// StockEmbedding_81131932221742
// MI455X (gfx1250) — matrix operand order repaired and checked by reading, not compiled, not run on hardware
//
#include <hip/hip_runtime.h>

// ---------------------------------------------------------------------------
// MI455X (gfx1250, wave32) implementation.
// Strategy: fold pos_table@fin_W and tech_W@fin_W on device (tiny kernels),
// reduce the per-token GEMM to K=100 (pad 128), run it as f16 WMMA
// (V_WMMA_F32_16X16X32_F16) with f32 accumulation. Main kernel is HBM-store
// bound (~131 MB output -> ~6us ideal at 23.3 TB/s), so the epilogue issues
// all P2 gathers before waiting and stores through immediate-offset pointers.
// ---------------------------------------------------------------------------

typedef _Float16 v16h __attribute__((ext_vector_type(16)));
typedef float    v8f  __attribute__((ext_vector_type(8)));
typedef unsigned int v4u __attribute__((ext_vector_type(4)));

#define B_    256
#define T_    1000
#define NTOK  (B_ * T_)
#define NTILE (NTOK / 16)      // 16000 tiles of 16 tokens
#define WIN   14
#define ALPHA (2.0f / 15.0f)

// workspace layout (in floats)
#define WS_P2    0                       // 1000*128  folded positional table
#define WS_TW2   (WS_P2 + 1000 * 128)    // 3*128     folded tech weights
#define WS_BIAS2 (WS_TW2 + 3 * 128)      // 128       folded bias
#define WS_FEAT  (WS_BIAS2 + 128)        // NTOK*8    per-token scan features

// ---------------- P2 = pos_table(1000x64) @ fin_W[33:97](64x128) -----------
__global__ void k_p2(const float* __restrict__ pos_table,
                     const float* __restrict__ fin_W,
                     float* __restrict__ ws) {
  int row = blockIdx.x;        // 0..999
  int n   = threadIdx.x;       // 0..127
  float acc = 0.f;
  for (int j = 0; j < 64; ++j)
    acc += pos_table[row * 64 + j] * fin_W[(33 + j) * 128 + n];
  ws[WS_P2 + row * 128 + n] = acc;
}

// ------- TW2 = tech_W(3x64) @ fin_W[161:225]; bias2 = fin_b + tech_b@... ---
__global__ void k_tw2(const float* __restrict__ tech_W,
                      const float* __restrict__ tech_b,
                      const float* __restrict__ fin_W,
                      const float* __restrict__ fin_b,
                      float* __restrict__ ws) {
  int n = threadIdx.x;         // 0..127
  for (int i = 0; i < 3; ++i) {
    float acc = 0.f;
    for (int j = 0; j < 64; ++j)
      acc += tech_W[i * 64 + j] * fin_W[(161 + j) * 128 + n];
    ws[WS_TW2 + i * 128 + n] = acc;
  }
  float bb = fin_b[n];
  for (int j = 0; j < 64; ++j)
    bb += tech_b[j] * fin_W[(161 + j) * 128 + n];
  ws[WS_BIAS2 + n] = bb;
}

// ------------- sequential per-row scans: vel/acc/sma/ema/rsi ---------------
__global__ void k_scan(const float* __restrict__ x, float* __restrict__ ws) {
  int bi = blockIdx.x * blockDim.x + threadIdx.x;   // batch row 0..255
  if (bi >= B_) return;
  const float* xr = x + (size_t)bi * T_ * 2;        // price at [2t], tval at [2t+1]
  float sp = 0.f, sg = 0.f, sl = 0.f, ema = 0.f, pprev = 0.f, dvprev = 0.f;
  for (int t = 0; t < T_; ++t) {
    float p  = xr[t * 2];
    float dv = (t >= 1) ? (p - pprev) : 0.f;        // dv[t-1]
    float vel = dv;                                 // vel[t]
    float acc = (t >= 2) ? (dv - dvprev) : 0.f;     // acc[t]
    // causal SMA(price), window 14, divide by 14 always
    sp += p;
    if (t >= WIN) sp -= xr[(t - WIN) * 2];
    float sma = sp * (1.f / WIN);
    // EMA scan
    ema = (t == 0) ? p : (ALPHA * p + (1.f - ALPHA) * ema);
    // RSI: gain/loss SMA over dv sequence, shifted by 1
    float rsi = 0.f;
    if (t >= 1) {
      sg += fmaxf(dv, 0.f);
      sl += fmaxf(-dv, 0.f);
      if (t >= WIN + 1) {       // evict gain/loss[t-15] = f(price[t-14]-price[t-15])
        float dvo = xr[(t - WIN) * 2] - xr[(t - WIN - 1) * 2];
        sg -= fmaxf(dvo, 0.f);
        sl -= fmaxf(-dvo, 0.f);
      }
      float rs = (sg * (1.f / WIN)) / (sl * (1.f / WIN) + 1e-7f);
      rsi = 100.f - 100.f / (1.f + rs);
    }
    size_t base = (size_t)WS_FEAT + ((size_t)bi * T_ + t) * 8;
    ws[base + 0] = vel;
    ws[base + 1] = acc;
    ws[base + 2] = sma;
    ws[base + 3] = ema;
    ws[base + 4] = rsi;
    if (t >= 1) dvprev = dv;
    pprev = p;
  }
}

// ----------------------- main fused WMMA GEMM ------------------------------
// Persistent: each block builds the folded 128(K,padded)x128(N) f16 weight
// matrix in LDS once, then loops over 16-token tiles.
__global__ void __launch_bounds__(256) k_gemm(
    const float* __restrict__ x, const int* __restrict__ time_steps,
    const float* __restrict__ w0p, const float* __restrict__ b0p,
    const float* __restrict__ w,   const float* __restrict__ bvec,
    const float* __restrict__ vel_W, const float* __restrict__ vel_b,
    const float* __restrict__ acc_W, const float* __restrict__ acc_b,
    const float* __restrict__ fin_W, const float* __restrict__ ws,
    float* __restrict__ out)
{
  __shared__ _Float16 Bsh[128 * 128];   // [n][k] (transposed for b128 frag loads)
  __shared__ _Float16 Ash[16 * 128];    // [m][k]
  __shared__ float    tok[16 * 8];      // per-token scalars
  __shared__ float    cp[192];          // w | b | vel_W | vel_b | acc_W | acc_b

  const int ti = threadIdx.x;
  const float w0 = w0p[0], b0 = b0p[0];
  const float* TW2   = ws + WS_TW2;
  const float* bias2 = ws + WS_BIAS2;
  const float* P2    = ws + WS_P2;
  const float* FT    = ws + WS_FEAT;

  if (ti < 32)       cp[ti] = w[ti];
  else if (ti < 64)  cp[ti] = bvec[ti - 32];
  else if (ti < 96)  cp[ti] = vel_W[ti - 64];
  else if (ti < 128) cp[ti] = vel_b[ti - 96];
  else if (ti < 160) cp[ti] = acc_W[ti - 128];
  else if (ti < 192) cp[ti] = acc_b[ti - 160];

  // Build folded weights B[n][k], k layout:
  //   0..32  time rows fin_W[0:33]
  //   33..64 vel  rows fin_W[97:129]
  //   65..96 acc  rows fin_W[129:161]
  //   97..99 TW2 rows; 100..127 zero pad
  for (int i = 0; i < 64; ++i) {
    int lin = ti + 256 * i;
    int n = lin >> 7, k = lin & 127;
    float v = 0.f;
    if (k < 33)       v = fin_W[k * 128 + n];
    else if (k < 65)  v = fin_W[(97  + (k - 33)) * 128 + n];
    else if (k < 97)  v = fin_W[(129 + (k - 65)) * 128 + n];
    else if (k < 100) v = TW2[(k - 97) * 128 + n];
    Bsh[n * 128 + k] = (_Float16)v;
  }
  __syncthreads();

  const int wv   = ti >> 5;        // wave 0..7 -> N-tile
  const int lane = ti & 31;
  const int hf   = lane >> 4;      // lane half (ISA A/B fragment layout)
  const int mr   = lane & 15;
  const int ncol = wv * 16 + mr;   // this lane's output column

  // per-wave LDS fragment bases (invariant across tiles)
  const _Float16* abase = &Ash[mr * 128 + hf * 8];     // + kc, +16 within chunk
  const _Float16* bbase = &Bsh[ncol * 128 + hf * 8];   // + kc, +16 within chunk
  const float bn = bias2[ncol];

  union Frag { v16h v; v4u q[2]; };

  for (int tile = blockIdx.x; tile < NTILE; tile += gridDim.x) {
    const int g0 = tile * 16;
    if (ti < 16) {
      int g = g0 + ti;
      size_t fb = (size_t)g * 8;
      tok[ti * 8 + 0] = FT[fb + 0];                 // vel
      tok[ti * 8 + 1] = FT[fb + 1];                 // acc
      tok[ti * 8 + 2] = FT[fb + 2];                 // sma
      tok[ti * 8 + 3] = FT[fb + 3];                 // ema
      tok[ti * 8 + 4] = FT[fb + 4];                 // rsi
      tok[ti * 8 + 5] = x[(size_t)g * 2 + 1];       // t value
      tok[ti * 8 + 6] = __int_as_float(time_steps[g]);
    }
    __syncthreads();

    // Build A[m][k] for 16 tokens (sines + ReLU embeddings + tech scalars)
    for (int i = 0; i < 8; ++i) {
      int lin = ti + 256 * i;
      int m = lin >> 7, k = lin & 127;
      const float* tk = tok + m * 8;
      float v = 0.f;
      if (k == 0)       v = w0 * tk[5] + b0;
      else if (k < 33)  { int j = k - 1;  v = sinf(tk[5] * cp[j] + cp[32 + j]); }
      else if (k < 65)  { int j = k - 33; v = fmaxf(tk[0] * cp[64 + j]  + cp[96 + j],  0.f); }
      else if (k < 97)  { int j = k - 65; v = fmaxf(tk[1] * cp[128 + j] + cp[160 + j], 0.f); }
      else if (k == 97) v = tk[2];
      else if (k == 98) v = tk[3];
      else if (k == 99) v = tk[4];
      Ash[m * 128 + k] = (_Float16)v;
    }
    __syncthreads();

    // Preload all fragments (8x ds_load_b128 per matrix), then 4 back-to-back
    // WMMAs. A 16x32 f16 frag: lanes 0-15 K{0-7,16-23}, lanes 16-31
    // K{8-15,24-31}. B 32x16 f16 frag: same K order as A.
    Frag af[4], bf[4];
#pragma unroll
    for (int c4 = 0; c4 < 4; ++c4) {
      const int kc = c4 * 32;
      af[c4].q[0] = *(const v4u*)(abase + kc);
      af[c4].q[1] = *(const v4u*)(abase + kc + 16);
      bf[c4].q[0] = *(const v4u*)(bbase + kc);
      bf[c4].q[1] = *(const v4u*)(bbase + kc + 16);
    }
    v8f c = {};
#pragma unroll
    for (int c4 = 0; c4 < 4; ++c4)
      c = __builtin_amdgcn_wmma_f32_16x16x32_f16(false, af[c4].v, false,
                                                 bf[c4].v, (short)0, c,
                                                 false, false);

    // Epilogue: batch the 8 time_step reads, then 8 P2 gathers (all loads in
    // flight before any wait), then 8 immediate-offset stores.
    // C layout: lane -> N = lane&15 ; VGPR r -> M = (lane>>4)*8 + r
    int tsv[8];
#pragma unroll
    for (int r = 0; r < 8; ++r)
      tsv[r] = __float_as_int(tok[(hf * 8 + r) * 8 + 6]);

    float p2v[8];
#pragma unroll
    for (int r = 0; r < 8; ++r)
      p2v[r] = P2[(size_t)tsv[r] * 128 + ncol];

    float* op = out + ((size_t)(g0 + hf * 8)) * 128 + ncol;
#pragma unroll
    for (int r = 0; r < 8; ++r)
      op[(size_t)r * 128] = c[r] + bn + p2v[r];

    __syncthreads();   // protect tok/Ash rebuild next iteration
  }
}

// ---------------------------------------------------------------------------
extern "C" void kernel_launch(void* const* d_in, const int* in_sizes, int n_in,
                              void* d_out, int out_size, void* d_ws, size_t ws_size,
                              hipStream_t stream) {
  const float* x          = (const float*)d_in[0];
  const int*   time_steps = (const int*)d_in[1];
  // d_in[2] technical_indicators: unused by the reference math
  const float* w0         = (const float*)d_in[3];
  const float* b0         = (const float*)d_in[4];
  const float* w          = (const float*)d_in[5];
  const float* b          = (const float*)d_in[6];
  const float* pos_table  = (const float*)d_in[7];
  const float* vel_W      = (const float*)d_in[8];
  const float* vel_b      = (const float*)d_in[9];
  const float* acc_W      = (const float*)d_in[10];
  const float* acc_b      = (const float*)d_in[11];
  const float* tech_W     = (const float*)d_in[12];
  const float* tech_b     = (const float*)d_in[13];
  const float* fin_W      = (const float*)d_in[14];
  const float* fin_b      = (const float*)d_in[15];
  float* out = (float*)d_out;
  float* ws  = (float*)d_ws;     // needs ~8.7 MB

  k_p2  <<<1000, 128, 0, stream>>>(pos_table, fin_W, ws);
  k_tw2 <<<1,    128, 0, stream>>>(tech_W, tech_b, fin_W, fin_b, ws);
  k_scan<<<2,    128, 0, stream>>>(x, ws);
  k_gemm<<<2000, 256, 0, stream>>>(x, time_steps, w0, b0, w, b,
                                   vel_W, vel_b, acc_W, acc_b, fin_W, ws, out);
}
